// QNetwork_72662256713877
// MI455X (gfx1250) — matrix operand order repaired and checked by reading, not compiled, not run on hardware
//
#include <hip/hip_runtime.h>

typedef __attribute__((ext_vector_type(16))) _Float16 v16h;
typedef __attribute__((ext_vector_type(8)))  _Float16 v8h;
typedef __attribute__((ext_vector_type(8)))  float    v8f;

union V16H { v16h v; v8h h[2]; };

#define NB  10
#define NC  3
#define NH  50
#define NW_G 101
#define NKPT 4096
#define GRID_N (NB*NC*NH*NW_G)

// ---------------- grid build: last-write-wins scatter via atomicMax ----------------

__global__ void order_init_kernel(int* __restrict__ order, int n) {
    int idx = blockIdx.x * blockDim.x + threadIdx.x;
    if (idx < n) order[idx] = (int)0x80000000;   // INT_MIN
}

__global__ void scatter_kernel(const int* __restrict__ x, int* __restrict__ order) {
    const int total = NB * NC * (NKPT + 1);
    int idx = blockIdx.x * blockDim.x + threadIdx.x;
    if (idx >= total) return;
    int k  = idx % (NKPT + 1);
    int bc = idx / (NKPT + 1);
    int col, row;
    if (k == 0) { col = 50; row = 48; }
    else {
        const int* p = x + ((size_t)bc * NKPT + (k - 1)) * 2;
        col = p[0]; row = p[1];
    }
    const int dr[10] = {-1,-1,-1, 0, 0, 0, 1, 1, 1, 0};
    const int dc[10] = {-1, 0, 1,-1, 0, 1,-1, 0, 1, 0};
#pragma unroll
    for (int j = 0; j < 10; ++j) {
        int rr = row + dr[j];
        int cc = col + dc[j];
        if (rr < 0 || rr >= NH || cc < 0 || cc >= NW_G) continue;
        atomicMax(&order[(bc * NH + rr) * NW_G + cc], k * 10 + j);
    }
}

__global__ void grid_finalize_kernel(const int* __restrict__ order, float* __restrict__ grid, int n) {
    int idx = blockIdx.x * blockDim.x + threadIdx.x;
    if (idx >= n) return;
    int o = order[idx];
    float v = 0.0f;
    if (o != (int)0x80000000) v = ((o % 10) == 9) ? 1.0f : 0.5f;
    grid[idx] = v;
}

// ---------------- weight convert + pad: f32 [N,K] -> f16 [Npad, Kpad] (zeros in pad) ----

__global__ void cvt_pad_kernel(const float* __restrict__ src, _Float16* __restrict__ dst,
                               int N, int K, int Kpad, int total) {
    int idx = blockIdx.x * blockDim.x + threadIdx.x;
    if (idx >= total) return;
    int kk = idx % Kpad;
    int nn = idx / Kpad;
    float v = (nn < N && kk < K) ? src[(size_t)nn * K + kk] : 0.0f;
    dst[idx] = (_Float16)v;
}

// ---------------- async global->LDS helpers (CDNA5 TDM-adjacent path, ASYNCcnt) ----

__device__ __forceinline__ void async_b128_to_lds(const _Float16* gbase, unsigned goff_bytes,
                                                  unsigned lds_addr) {
    // GVS mode: mem = SGPR base + VGPR offset ; LDS dest addr in VGPR
    asm volatile("global_load_async_to_lds_b128 %0, %1, %2"
                 :: "v"(lds_addr), "v"(goff_bytes), "s"(gbase)
                 : "memory");
}

__device__ __forceinline__ void wait_asynccnt0() {
    asm volatile("s_wait_asynccnt 0x0" ::: "memory");
}

// ---------------- implicit-GEMM convolution with WMMA, LDS-staged tiles ----------------
// GEMM view: M = NB*OH*OW (16 rows/block), N = OC (one 16-col tile/wave), K = IC*KH*KW.
// A tile: im2col gather -> ds_store (needs zero halo). B tile: pure async DMA from the
// pre-converted, pre-padded f16 weight buffer. 64 K-values staged per barrier.

template<int IC, int IH, int IW, int OC, int OH, int OW, int KH, int KW, int ST, int PD>
__global__ __launch_bounds__(32 * (OC / 16)) void conv_wmma_kernel(
        const float* __restrict__ in, const _Float16* __restrict__ wh,
        const float* __restrict__ bias, float* __restrict__ out) {
    constexpr int M    = NB * OH * OW;
    constexpr int K    = IC * KH * KW;
    constexpr int KPAD = ((K + 63) / 64) * 64;
    constexpr int NWAVES = OC / 16;
    constexpr int KT   = 64;
    constexpr int NTHR = 32 * NWAVES;

    __shared__ __align__(16) _Float16 As[16 * KT];   // [row][k]
    __shared__ __align__(16) _Float16 Bs[OC * KT];   // [n][k]

    const int tid  = threadIdx.x;
    const int lane = tid & 31;
    const int wv   = tid >> 5;
    const int r    = lane & 15;
    const int hf   = lane >> 4;
    const int m_base = blockIdx.x * 16;
    const int n = wv * 16 + r;

    v8f acc = {};
    for (int kb = 0; kb < K; kb += KT) {
        // ---- B tile: async DMA, 8 halfs (16B) per issue, no guards (buffer padded) ----
        for (int i = tid; i < (OC * KT) / 8; i += NTHR) {
            int nn = i / (KT / 8);
            int kl = (i % (KT / 8)) * 8;
            unsigned lds  = (unsigned)(size_t)(Bs + nn * KT + kl);
            unsigned goff = (unsigned)(((size_t)nn * KPAD + kb + kl) * sizeof(_Float16));
            async_b128_to_lds(wh, goff, lds);
        }
        // ---- A tile: im2col gather (zero halo), f32 -> f16 ----
        for (int i = tid; i < 16 * KT; i += NTHR) {
            int row = i / KT;
            int kk  = kb + (i % KT);
            float val = 0.0f;
            int m = m_base + row;
            if (m < M && kk < K) {
                int ow = m % OW;  int t  = m / OW;
                int oh = t % OH;  int b  = t / OH;
                int kw = kk % KW; int t2 = kk / KW;
                int kh = t2 % KH; int ci = t2 / KH;
                int ih = oh * ST - PD + kh;
                int iw = ow * ST - PD + kw;
                if (ih >= 0 && ih < IH && iw >= 0 && iw < IW)
                    val = in[(((size_t)b * IC + ci) * IH + ih) * IW + iw];
            }
            As[i] = (_Float16)val;
        }
        wait_asynccnt0();
        __syncthreads();
        // ---- two 16x16x32 WMMA chunks, fragments via ds_load_b128 ----
#pragma unroll
        for (int c = 0; c < 2; ++c) {
            const _Float16* ap = As + r * KT + c * 32;
            const _Float16* bp = Bs + n * KT + c * 32;
            V16H a, bf;
            a.h[0]  = *(const v8h*)(ap + hf * 8);
            a.h[1]  = *(const v8h*)(ap + 16 + hf * 8);
            bf.h[0] = *(const v8h*)(bp + hf * 8);
            bf.h[1] = *(const v8h*)(bp + 16 + hf * 8);
            acc = __builtin_amdgcn_wmma_f32_16x16x32_f16(false, a.v, false, bf.v,
                                                         (short)0, acc, false, false);
        }
        __syncthreads();
    }

    const float bn = bias[n];
#pragma unroll
    for (int v = 0; v < 8; ++v) {
        const int m = m_base + v + 8 * hf;
        if (m < M) {
            int ow = m % OW; int t = m / OW;
            int oh = t % OH; int b = t / OH;
            float val = acc[v] + bn;
            val = val > 0.0f ? val : 0.0f;
            out[(((size_t)b * OC + n) * OH + oh) * OW + ow] = val;
        }
    }
}

// ---------------- maxpool (VALID) ----------------

__global__ void maxpool_kernel(const float* __restrict__ in, float* __restrict__ out,
                               int BC, int IH, int IW, int k, int s, int OH, int OW) {
    const int total = BC * OH * OW;
    int idx = blockIdx.x * blockDim.x + threadIdx.x;
    if (idx >= total) return;
    int ow = idx % OW; int t = idx / OW;
    int oh = t % OH;  int bc = t / OH;
    float m = -3.402823466e38f;
    for (int i = 0; i < k; ++i)
        for (int j = 0; j < k; ++j) {
            float v = in[((size_t)bc * IH + oh * s + i) * IW + ow * s + j];
            m = v > m ? v : m;
        }
    out[idx] = m;
}

// ---------------- global average pool [B,C,H,W] -> [B,C] ----------------

__global__ void mean_kernel(const float* __restrict__ in, float* __restrict__ out,
                            int BC, int HW) {
    int idx = blockIdx.x * blockDim.x + threadIdx.x;
    if (idx >= BC) return;
    const float* p = in + (size_t)idx * HW;
    float s = 0.0f;
    for (int i = 0; i < HW; ++i) s += p[i];
    out[idx] = s / (float)HW;
}

// ---------------- FC as WMMA GEMM (LDS-staged, async B): out = act(A.W^T + b) ----------

template<int N, int K, int RELU>
__global__ __launch_bounds__(32 * ((N + 15) / 16)) void fc_wmma_kernel(
        const float* __restrict__ A, const _Float16* __restrict__ Wh,
        const float* __restrict__ bias, float* __restrict__ out) {
    constexpr int M    = NB;
    constexpr int NR   = ((N + 15) / 16) * 16;
    constexpr int KPAD = ((K + 63) / 64) * 64;
    constexpr int NWAVES = NR / 16;
    constexpr int KT   = 64;
    constexpr int NTHR = 32 * NWAVES;

    __shared__ __align__(16) _Float16 As[16 * KT];
    __shared__ __align__(16) _Float16 Bs[NR * KT];

    const int tid  = threadIdx.x;
    const int lane = tid & 31;
    const int wv   = tid >> 5;
    const int r    = lane & 15;
    const int hf   = lane >> 4;
    const int n = wv * 16 + r;

    v8f acc = {};
    for (int kb = 0; kb < K; kb += KT) {
        for (int i = tid; i < (NR * KT) / 8; i += NTHR) {
            int nn = i / (KT / 8);
            int kl = (i % (KT / 8)) * 8;
            unsigned lds  = (unsigned)(size_t)(Bs + nn * KT + kl);
            unsigned goff = (unsigned)(((size_t)nn * KPAD + kb + kl) * sizeof(_Float16));
            async_b128_to_lds(Wh, goff, lds);
        }
        for (int i = tid; i < 16 * KT; i += NTHR) {
            int row = i / KT;
            int kk  = kb + (i % KT);
            As[i] = (_Float16)((row < M && kk < K) ? A[(size_t)row * K + kk] : 0.0f);
        }
        wait_asynccnt0();
        __syncthreads();
#pragma unroll
        for (int c = 0; c < 2; ++c) {
            const _Float16* ap = As + r * KT + c * 32;
            const _Float16* bp = Bs + n * KT + c * 32;
            V16H a, bf;
            a.h[0]  = *(const v8h*)(ap + hf * 8);
            a.h[1]  = *(const v8h*)(ap + 16 + hf * 8);
            bf.h[0] = *(const v8h*)(bp + hf * 8);
            bf.h[1] = *(const v8h*)(bp + 16 + hf * 8);
            acc = __builtin_amdgcn_wmma_f32_16x16x32_f16(false, a.v, false, bf.v,
                                                         (short)0, acc, false, false);
        }
        __syncthreads();
    }

    const float bn = (n < N) ? bias[n] : 0.0f;
#pragma unroll
    for (int v = 0; v < 8; ++v) {
        const int m = v + 8 * hf;
        if (m < M && n < N) {
            float val = acc[v] + bn;
            if (RELU) val = val > 0.0f ? val : 0.0f;
            out[(size_t)m * N + n] = val;
        }
    }
}

// ---------------- host side ----------------

static inline int cdiv(int a, int b) { return (a + b - 1) / b; }

extern "C" void kernel_launch(void* const* d_in, const int* in_sizes, int n_in,
                              void* d_out, int out_size, void* d_ws, size_t ws_size,
                              hipStream_t stream) {
    (void)in_sizes; (void)n_in; (void)out_size; (void)ws_size;
    const int*   x   = (const int*)d_in[0];
    const float* w1  = (const float*)d_in[1];
    const float* b1  = (const float*)d_in[2];
    const float* w2  = (const float*)d_in[3];
    const float* b2  = (const float*)d_in[4];
    const float* w3  = (const float*)d_in[5];
    const float* b3  = (const float*)d_in[6];
    const float* wl1 = (const float*)d_in[7];
    const float* bl1 = (const float*)d_in[8];
    const float* wl2 = (const float*)d_in[9];
    const float* bl2 = (const float*)d_in[10];
    float* out = (float*)d_out;

    char* ws = (char*)d_ws;
    size_t off = 0;
    auto carve = [&](size_t bytes) -> void* {
        void* p = ws + off;
        off = (off + bytes + 255) & ~(size_t)255;
        return p;
    };

    int*   order = (int*)  carve((size_t)GRID_N * 4);
    float* grid  = (float*)carve((size_t)GRID_N * 4);
    float* c1    = (float*)carve((size_t)NB * 32 * 25 * 51 * 4);
    float* p1    = (float*)carve((size_t)NB * 32 * 12 * 25 * 4);
    float* c2    = (float*)carve((size_t)NB * 64 * 12 * 25 * 4);
    float* p2    = (float*)carve((size_t)NB * 64 * 6 * 12 * 4);
    float* c3    = (float*)carve((size_t)NB * 128 * 6 * 12 * 4);
    float* p3    = (float*)carve((size_t)NB * 128 * 3 * 6 * 4);
    float* mn    = (float*)carve((size_t)NB * 128 * 4);
    float* f1    = (float*)carve((size_t)NB * 128 * 4);
    // padded f16 weight buffers (Npad x Kpad)
    _Float16* w1h = (_Float16*)carve((size_t)32 * 192 * 2);     // K=147 -> 192
    _Float16* w2h = (_Float16*)carve((size_t)64 * 832 * 2);     // K=800 -> 832
    _Float16* w3h = (_Float16*)carve((size_t)128 * 576 * 2);    // K=576
    _Float16* l1h = (_Float16*)carve((size_t)128 * 128 * 2);    // K=128
    _Float16* l2h = (_Float16*)carve((size_t)16 * 128 * 2);     // N=5 -> 16

    // 0) convert + pad weights to f16 (cheap; must be redone every call)
    cvt_pad_kernel<<<cdiv(32 * 192, 256), 256, 0, stream>>>(w1, w1h, 32, 147, 192, 32 * 192);
    cvt_pad_kernel<<<cdiv(64 * 832, 256), 256, 0, stream>>>(w2, w2h, 64, 800, 832, 64 * 832);
    cvt_pad_kernel<<<cdiv(128 * 576, 256), 256, 0, stream>>>(w3, w3h, 128, 576, 576, 128 * 576);
    cvt_pad_kernel<<<cdiv(128 * 128, 256), 256, 0, stream>>>(wl1, l1h, 128, 128, 128, 128 * 128);
    cvt_pad_kernel<<<cdiv(16 * 128, 256), 256, 0, stream>>>(wl2, l2h, 5, 128, 128, 16 * 128);

    // 1) build grid
    order_init_kernel<<<cdiv(GRID_N, 256), 256, 0, stream>>>(order, GRID_N);
    scatter_kernel<<<cdiv(NB * NC * (NKPT + 1), 256), 256, 0, stream>>>(x, order);
    grid_finalize_kernel<<<cdiv(GRID_N, 256), 256, 0, stream>>>(order, grid, GRID_N);

    // 2) conv1 (7x7 s2 p3) + relu : [10,3,50,101] -> [10,32,25,51]
    conv_wmma_kernel<3, 50, 101, 32, 25, 51, 7, 7, 2, 3>
        <<<cdiv(NB * 25 * 51, 16), 64, 0, stream>>>(grid, w1h, b1, c1);
    maxpool_kernel<<<cdiv(NB * 32 * 12 * 25, 256), 256, 0, stream>>>(
        c1, p1, NB * 32, 25, 51, 3, 2, 12, 25);

    // 3) conv2 (5x5 s1 p2) + relu : [10,32,12,25] -> [10,64,12,25]
    conv_wmma_kernel<32, 12, 25, 64, 12, 25, 5, 5, 1, 2>
        <<<cdiv(NB * 12 * 25, 16), 128, 0, stream>>>(p1, w2h, b2, c2);
    maxpool_kernel<<<cdiv(NB * 64 * 6 * 12, 256), 256, 0, stream>>>(
        c2, p2, NB * 64, 12, 25, 2, 2, 6, 12);

    // 4) conv3 (3x3 s1 p1) + relu : [10,64,6,12] -> [10,128,6,12]
    conv_wmma_kernel<64, 6, 12, 128, 6, 12, 3, 3, 1, 1>
        <<<cdiv(NB * 6 * 12, 16), 256, 0, stream>>>(p2, w3h, b3, c3);
    maxpool_kernel<<<cdiv(NB * 128 * 3 * 6, 256), 256, 0, stream>>>(
        c3, p3, NB * 128, 6, 12, 2, 2, 3, 6);

    // 5) global mean : -> [10,128]
    mean_kernel<<<cdiv(NB * 128, 256), 256, 0, stream>>>(p3, mn, NB * 128, 3 * 6);

    // 6) fc1 (relu) : [10,128] x [128,128]^T -> [10,128]
    fc_wmma_kernel<128, 128, 1><<<1, 256, 0, stream>>>(mn, l1h, bl1, f1);
    // 7) fc2 : [10,128] x [5,128]^T -> [10,5]  (d_out)
    fc_wmma_kernel<5, 128, 0><<<1, 32, 0, stream>>>(f1, l2h, bl2, out);
}
